// CNEN_39779987096265
// MI455X (gfx1250) — matrix operand order repaired and checked by reading, not compiled, not run on hardware
//
#include <hip/hip_runtime.h>
#include <stdint.h>

typedef __bf16 bf16;
typedef __attribute__((ext_vector_type(16))) __bf16 v16bf;
typedef __attribute__((ext_vector_type(8)))  __bf16 v8bf;
typedef __attribute__((ext_vector_type(8)))  float  v8f;

#define NN      4096
#define DIM_IN  256
#define DIM_H   128
#define DIM_OUT 128

__device__ __forceinline__ bf16 f2bf(float x) {
  unsigned u = __builtin_bit_cast(unsigned, x);
  unsigned r = u + 0x7FFFu + ((u >> 16) & 1u);
  unsigned short h = (unsigned short)(r >> 16);
  return __builtin_bit_cast(bf16, h);
}

// Async 16B global -> LDS copy (per-lane), GVS addressing: saddr + vgpr offset.
// Tracked by ASYNCcnt.
__device__ __forceinline__ void async_copy16(uint32_t lds_byte_addr,
                                             const void* sbase,
                                             int32_t vbyte_off) {
  asm volatile("global_load_async_to_lds_b128 %0, %1, %2"
               :: "v"(lds_byte_addr), "v"(vbyte_off), "s"(sbase)
               : "memory");
}

// fp32 [R,C] -> bf16 row-major and/or bf16 transposed [C,R]
template <bool RM, bool TR>
__global__ void convert_kernel(const float* __restrict__ src,
                               bf16* __restrict__ rm,
                               bf16* __restrict__ tr,
                               int R, int C) {
  long i = (long)blockIdx.x * blockDim.x + threadIdx.x;
  long n = (long)R * C;
  if (i >= n) return;
  int r = (int)(i / C);
  int c = (int)(i % C);
  bf16 v = f2bf(src[i]);
  if (RM) rm[i] = v;
  if (TR) tr[(long)c * R + r] = v;
}

// C[M,N] = f(A[M,K] @ B[K,N]) with B supplied transposed: BT[N,K] row-major.
// MODE: 0 = identity, 1 = relu, 2 = exp.
// Outputs (compile-time): OF -> Cf fp32 [M,N], OB -> Cb bf16 [M,N], OT -> Ct bf16 [N,M].
// Requires: M,N multiples of 128; K multiple of 32 (and >= 64 for the peel).
// Block: 256 threads (8 waves). Block tile 128x128; wave tile 32x64 (2x4 WMMA).
// K-step 32. A/B 128x32 tiles staged in LDS via async copies, double-buffered;
// last K-step peeled so the steady-state loop is branch-free.
template <int MODE, bool OF, bool OB, bool OT>
__global__ __launch_bounds__(256)
void gemm_bf16_kernel(const bf16* __restrict__ A,
                      const bf16* __restrict__ BT,
                      float* __restrict__ Cf,
                      bf16* __restrict__ Cb,
                      bf16* __restrict__ Ct,
                      int M, int N, int K) {
  constexpr int LROW  = 40;          // row stride in elems (80B: pad kills bank conflicts)
  constexpr int LTILE = 128 * LROW;  // elems per 128x32 tile
  __shared__ bf16 smem[2][2][LTILE]; // [buffer][A/B][tile]  = 40 KB

  const int tid  = threadIdx.x;
  const int wave = tid >> 5;
  const int lane = tid & 31;
  const int half = lane >> 4;        // 0: lanes 0-15, 1: lanes 16-31
  const int l    = lane & 15;

  const int row0 = blockIdx.y * 128;           // block M origin
  const int col0 = blockIdx.x * 128;           // block N origin
  const int wrow = (wave >> 1) * 32;           // wave M offset in tile
  const int wcol = (wave & 1) * 64;            // wave N offset in tile

  const uint32_t smem_base = (uint32_t)(uintptr_t)(&smem[0][0][0]);

  // Copy mapping: 512 chunks of 16B per tile; thread t does chunks t and t+256.
  // chunk -> row = chunk>>2, 16B-piece o = chunk&3.
  auto issue_stage = [&](int buf, int k0) {
#pragma unroll
    for (int q = 0; q < 2; ++q) {
      const int chunk = tid + q * 256;
      const int r = chunk >> 2;
      const int o = chunk & 3;
      const uint32_t lA = smem_base +
          (uint32_t)(((buf * 2 + 0) * LTILE + r * LROW + o * 8) * 2);
      const int32_t gA = (int32_t)(((row0 + r) * K + k0 + o * 8) * 2);
      async_copy16(lA, A, gA);
      const uint32_t lB = smem_base +
          (uint32_t)(((buf * 2 + 1) * LTILE + r * LROW + o * 8) * 2);
      const int32_t gB = (int32_t)(((col0 + r) * K + k0 + o * 8) * 2);
      async_copy16(lB, BT, gB);
    }
  };

  v8f acc[2][4] = {};

  auto compute_stage = [&](int buf) {
    const bf16* sA = &smem[buf][0][0];
    const bf16* sB = &smem[buf][1][0];

    v16bf afrag[2], bfrag[4];
#pragma unroll
    for (int i = 0; i < 2; ++i) {
      // A frag (16x32): lane<16 -> M=l, K {0..7, 16..23}; lane>=16 -> K {8..15, 24..31}
      const bf16* a = sA + (wrow + i * 16 + l) * LROW + half * 8;
      v8bf alo = *(const v8bf*)a;
      v8bf ahi = *(const v8bf*)(a + 16);
      afrag[i] = __builtin_shufflevector(alo, ahi,
                   0, 1, 2, 3, 4, 5, 6, 7, 8, 9, 10, 11, 12, 13, 14, 15);
    }
#pragma unroll
    for (int j = 0; j < 4; ++j) {
      // B frag (32x16): lane<16 -> N=l, K {0..7, 16..23}; lane>=16 -> N=l, K {8..15, 24..31}
      const bf16* b = sB + (wcol + j * 16 + l) * LROW + half * 8;
      v8bf blo = *(const v8bf*)b;
      v8bf bhi = *(const v8bf*)(b + 16);
      bfrag[j] = __builtin_shufflevector(blo, bhi,
                   0, 1, 2, 3, 4, 5, 6, 7, 8, 9, 10, 11, 12, 13, 14, 15);
    }
#pragma unroll
    for (int i = 0; i < 2; ++i)
#pragma unroll
      for (int j = 0; j < 4; ++j)
        acc[i][j] = __builtin_amdgcn_wmma_f32_16x16x32_bf16(
            false, afrag[i], false, bfrag[j], (short)0, acc[i][j],
            false, false);
  };

  // Prologue: stage 0 in flight.
  issue_stage(0, 0);
  int buf = 0;
  // Steady state: branch-free double-buffered pipeline.
  for (int k0 = 0; k0 + 32 < K; k0 += 32) {
    issue_stage(buf ^ 1, k0 + 32);
    asm volatile("s_wait_asynccnt 0x4" ::: "memory");  // stage 'buf' landed
    __syncthreads();                                   // ... in every wave
    compute_stage(buf);
    __syncthreads();                                   // guard LDS buffer reuse
    buf ^= 1;
  }
  // Peeled last K-step: nothing left to prefetch.
  asm volatile("s_wait_asynccnt 0x0" ::: "memory");
  __syncthreads();
  compute_stage(buf);

  // Epilogue. C/D layout: VGPR e -> M = e + 8*half, N = l (within 16x16 tile).
#pragma unroll
  for (int i = 0; i < 2; ++i) {
#pragma unroll
    for (int j = 0; j < 4; ++j) {
      float x[8];
#pragma unroll
      for (int e = 0; e < 8; ++e) {
        float v = acc[i][j][e];
        if (MODE == 1) v = fmaxf(v, 0.0f);
        if (MODE == 2) v = __expf(v);
        x[e] = v;
      }
      const size_t rbase = (size_t)(row0 + wrow + i * 16 + half * 8);
      const size_t c     = (size_t)(col0 + wcol + j * 16 + l);
      if (OF) {
#pragma unroll
        for (int e = 0; e < 8; ++e)
          Cf[(rbase + e) * (size_t)N + c] = x[e];
      }
      if (OB) {
#pragma unroll
        for (int e = 0; e < 8; ++e)
          Cb[(rbase + e) * (size_t)N + c] = f2bf(x[e]);
      }
      if (OT) {
        // 8 consecutive rows at one column -> contiguous 16B in Ct[N,M].
        v8bf v;
#pragma unroll
        for (int e = 0; e < 8; ++e) v[e] = f2bf(x[e]);
        *(v8bf*)(Ct + c * (size_t)M + rbase) = v;
      }
    }
  }
}

static void conv(hipStream_t s, const float* src, bf16* rm, bf16* tr, int R, int C) {
  long n = (long)R * C;
  int blocks = (int)((n + 255) / 256);
  if (rm && tr)
    convert_kernel<true, true><<<blocks, 256, 0, s>>>(src, rm, tr, R, C);
  else if (rm)
    convert_kernel<true, false><<<blocks, 256, 0, s>>>(src, rm, tr, R, C);
  else
    convert_kernel<false, true><<<blocks, 256, 0, s>>>(src, rm, tr, R, C);
}

static void gemm(hipStream_t s, const bf16* A, const bf16* BT,
                 float* Cf, bf16* Cb, bf16* Ct, int M, int N, int K, int mode) {
  dim3 grid(N / 128, M / 128), block(256);
  if (mode == 0 && Cf)                 // fp32 out only (final projections)
    gemm_bf16_kernel<0, true, false, false><<<grid, block, 0, s>>>(A, BT, Cf, Cb, Ct, M, N, K);
  else if (mode == 0)                  // bf16 row-major out only
    gemm_bf16_kernel<0, false, true, false><<<grid, block, 0, s>>>(A, BT, Cf, Cb, Ct, M, N, K);
  else if (mode == 1 && Cb)            // relu, bf16 rm + transposed
    gemm_bf16_kernel<1, false, true, true><<<grid, block, 0, s>>>(A, BT, Cf, Cb, Ct, M, N, K);
  else if (mode == 1)                  // relu, transposed only
    gemm_bf16_kernel<1, false, false, true><<<grid, block, 0, s>>>(A, BT, Cf, Cb, Ct, M, N, K);
  else                                 // exp, fp32 + bf16 rm (S matrix)
    gemm_bf16_kernel<2, true, true, false><<<grid, block, 0, s>>>(A, BT, Cf, Cb, Ct, M, N, K);
}

extern "C" void kernel_launch(void* const* d_in, const int* in_sizes, int n_in,
                              void* d_out, int out_size, void* d_ws, size_t ws_size,
                              hipStream_t stream) {
  (void)in_sizes; (void)n_in; (void)out_size; (void)ws_size;
  const float* A_s = (const float*)d_in[0];
  const float* X_s = (const float*)d_in[1];
  const float* A_t = (const float*)d_in[2];
  const float* X_t = (const float*)d_in[3];
  const float* W1  = (const float*)d_in[4];
  const float* W2  = (const float*)d_in[5];
  const float* W3  = (const float*)d_in[6];
  const float* W4  = (const float*)d_in[7];

  float* out_s = (float*)d_out;                          // [N, DIM_OUT]
  float* out_t = out_s + (size_t)NN * DIM_OUT;           // [N, DIM_OUT]
  float* S_out = out_t + (size_t)NN * DIM_OUT;           // [N, N]

  // Workspace carve (256B aligned slots).
  char* p = (char*)d_ws;
  auto carve = [&](size_t bytes) -> char* {
    char* r = p;
    p += (bytes + 255) & ~(size_t)255;
    return r;
  };
  bf16* Abf   = (bf16*)carve((size_t)NN * NN * 2);          // A_s then A_t (reused)
  bf16* Sbf   = (bf16*)carve((size_t)NN * NN * 2);          // S in bf16
  bf16* XsT   = (bf16*)carve((size_t)DIM_IN * NN * 2);      // X_s^T
  bf16* XtT   = (bf16*)carve((size_t)DIM_IN * NN * 2);      // X_t^T
  bf16* W1T   = (bf16*)carve((size_t)DIM_H * DIM_IN * 2);
  bf16* W2T   = (bf16*)carve((size_t)DIM_H * DIM_H * 2);
  bf16* W3T   = (bf16*)carve((size_t)DIM_H * DIM_H * 2);
  bf16* W4T   = (bf16*)carve((size_t)DIM_OUT * DIM_H * 2);
  bf16* Hbuf  = (bf16*)carve((size_t)NN * DIM_IN * 2);      // A@X
  bf16* V1T   = (bf16*)carve((size_t)DIM_H * NN * 2);       // layer-1 act, transposed
  bf16* H2    = (bf16*)carve((size_t)NN * DIM_H * 2);       // A@act1
  bf16* Xs2rm = (bf16*)carve((size_t)NN * DIM_H * 2);
  bf16* Xs2T  = (bf16*)carve((size_t)DIM_H * NN * 2);
  bf16* Xt2rm = (bf16*)carve((size_t)NN * DIM_H * 2);
  bf16* Xt2T  = (bf16*)carve((size_t)DIM_H * NN * 2);
  bf16* Prm   = (bf16*)carve((size_t)NN * DIM_H * 2);       // Xs2 @ W3
  bf16* Us    = (bf16*)carve((size_t)NN * DIM_H * 2);       // S @ Xs2
  bf16* Ut    = (bf16*)carve((size_t)NN * DIM_H * 2);       // S @ Xt2

  // --- Precision conversions ---
  conv(stream, W1, nullptr, W1T, DIM_IN, DIM_H);   // W1T [DIM_H, DIM_IN]
  conv(stream, W2, nullptr, W2T, DIM_H, DIM_H);
  conv(stream, W3, nullptr, W3T, DIM_H, DIM_H);
  conv(stream, W4, nullptr, W4T, DIM_H, DIM_OUT);
  conv(stream, X_s, nullptr, XsT, NN, DIM_IN);     // XsT [DIM_IN, N]
  conv(stream, X_t, nullptr, XtT, NN, DIM_IN);

  // --- Source graph: two GCN layers ---
  conv(stream, A_s, Abf, nullptr, NN, NN);
  gemm(stream, Abf, XsT, nullptr, Hbuf, nullptr, NN, DIM_IN, NN, 0);       // H = A_s @ X_s
  gemm(stream, Hbuf, W1T, nullptr, nullptr, V1T, NN, DIM_H, DIM_IN, 1);    // act1^T = relu(H @ W1)^T
  gemm(stream, Abf, V1T, nullptr, H2, nullptr, NN, DIM_H, NN, 0);          // H2 = A_s @ act1
  gemm(stream, H2, W2T, nullptr, Xs2rm, Xs2T, NN, DIM_H, DIM_H, 1);        // Xs = relu(H2 @ W2)

  // --- Target graph (reuse Abf, V1T, Hbuf, H2) ---
  conv(stream, A_t, Abf, nullptr, NN, NN);
  gemm(stream, Abf, XtT, nullptr, Hbuf, nullptr, NN, DIM_IN, NN, 0);
  gemm(stream, Hbuf, W1T, nullptr, nullptr, V1T, NN, DIM_H, DIM_IN, 1);
  gemm(stream, Abf, V1T, nullptr, H2, nullptr, NN, DIM_H, NN, 0);
  gemm(stream, H2, W2T, nullptr, Xt2rm, Xt2T, NN, DIM_H, DIM_H, 1);        // Xt

  // --- Attention scores: S = exp((Xs @ W3) @ Xt^T) ---
  gemm(stream, Xs2rm, W3T, nullptr, Prm, nullptr, NN, DIM_H, DIM_H, 0);    // P = Xs @ W3
  // B = Xt^T  =>  BT = Xt (row-major) directly.
  gemm(stream, Prm, Xt2rm, S_out, Sbf, nullptr, NN, NN, DIM_H, 2);         // S (fp32 out + bf16 copy)

  // --- Aggregation + output projection ---
  gemm(stream, Sbf, Xs2T, nullptr, Us, nullptr, NN, DIM_H, NN, 0);         // U_s = S @ Xs
  gemm(stream, Sbf, Xt2T, nullptr, Ut, nullptr, NN, DIM_H, NN, 0);         // U_t = S @ Xt
  gemm(stream, Us, W4T, out_s, nullptr, nullptr, NN, DIM_OUT, DIM_H, 0);   // out_s
  gemm(stream, Ut, W4T, out_t, nullptr, nullptr, NN, DIM_OUT, DIM_H, 0);   // out_t
}
